// GeometryClassifier_44753559224467
// MI455X (gfx1250) — matrix operand order repaired and checked by reading, not compiled, not run on hardware
//
#include <hip/hip_runtime.h>
#include <hip/hip_bf16.h>
#include <math.h>

#define NN 50000
#define NE 800000
#define NG 64

typedef __attribute__((ext_vector_type(16))) _Float16 v16h;
typedef __attribute__((ext_vector_type(8)))  _Float16 v8h;
typedef __attribute__((ext_vector_type(4)))  _Float16 v4h;
typedef __attribute__((ext_vector_type(8)))  float    v8f;

// ---------------- degree / dinv ----------------
__global__ void deg_kernel(const int* __restrict__ dst, float* __restrict__ deg) {
    int e = blockIdx.x * blockDim.x + threadIdx.x;
    if (e < NE) atomicAdd(&deg[dst[e]], 1.0f);
}

__global__ void dinv_kernel(const float* __restrict__ deg, float* __restrict__ dinv) {
    int n = blockIdx.x * blockDim.x + threadIdx.x;
    if (n < NN) dinv[n] = rsqrtf(deg[n] + 1.0f);   // GCN: self-loop adds 1
}

// ---------------- GCN1: aggregate 2-ch x, then [N,2]@[2,64] ----------------
__global__ void gcn1_scatter(const float* __restrict__ x, const int* __restrict__ src,
                             const int* __restrict__ dst, const float* __restrict__ dinv,
                             float* __restrict__ aggx) {
    int idx = blockIdx.x * blockDim.x + threadIdx.x;
    if (idx >= 2 * NE) return;
    int e = idx >> 1, c = idx & 1;
    int s = src[e], d = dst[e];
    float nrm = dinv[s] * dinv[d];
    atomicAdd(&aggx[d * 2 + c], x[s * 2 + c] * nrm);
}

__global__ void gcn1_fin(const float* __restrict__ x, const float* __restrict__ aggx,
                         const float* __restrict__ dinv, const float* __restrict__ W,
                         const float* __restrict__ b, float* __restrict__ h1) {
    int idx = blockIdx.x * blockDim.x + threadIdx.x;
    if (idx >= NN * 64) return;
    int n = idx >> 6, c = idx & 63;
    float di = dinv[n], d2 = di * di;
    float z0 = aggx[n * 2 + 0] + x[n * 2 + 0] * d2;
    float z1 = aggx[n * 2 + 1] + x[n * 2 + 1] * d2;
    float v = z0 * W[c] + z1 * W[64 + c] + b[c];
    h1[idx] = v > 0.f ? v : 0.f;
}

// ---------------- SAGE: mean-agg + root + residual ----------------
__global__ void sage_scatter(const float* __restrict__ h1, const int* __restrict__ src,
                             const int* __restrict__ dst, float* __restrict__ s) {
    int idx = blockIdx.x * blockDim.x + threadIdx.x;
    if (idx >= NE * 64) return;
    int e = idx >> 6, c = idx & 63;
    atomicAdd(&s[dst[e] * 64 + c], h1[src[e] * 64 + c]);
}

__global__ void sage_fin(const float* __restrict__ h1, const float* __restrict__ s,
                         const float* __restrict__ deg, const float* __restrict__ Wl,
                         const float* __restrict__ Wr, const float* __restrict__ Wres,
                         const float* __restrict__ bs, const float* __restrict__ bres,
                         float* __restrict__ h2) {
    __shared__ float sm[64], sx[64];
    int n = blockIdx.x, c = threadIdx.x;           // 128 threads
    if (c < 64) {
        float d = deg[n]; d = d > 1.f ? d : 1.f;
        sm[c] = s[n * 64 + c] / d;
        sx[c] = h1[n * 64 + c];
    }
    __syncthreads();
    float acc = bs[c] + bres[c];
    #pragma unroll 8
    for (int k = 0; k < 64; ++k)
        acc += sm[k] * Wl[k * 128 + c] + sx[k] * (Wr[k * 128 + c] + Wres[k * 128 + c]);
    h2[n * 128 + c] = acc > 0.f ? acc : 0.f;
}

// ---------------- weight f32 -> f16 conversion, pre-swizzled into WMMA B-fragment order --
// Fragment (j = n-tile, k = k-tile) holds 32 lanes x 16 halfs contiguously:
//   packed[frag*512 + lane*16 + i] = W[(k*32 + (i>>3)*16 + (lane>>4)*8 + (i&7))*128 + (j*16 + (lane&15))]
__global__ void cvt_w(const float* __restrict__ W1, const float* __restrict__ W2,
                      _Float16* __restrict__ W1p, _Float16* __restrict__ W2p) {
    int idx = blockIdx.x * blockDim.x + threadIdx.x;   // 32768 threads
    {
        int frag = idx >> 9, t = idx & 511;
        int lane = t >> 4, i = t & 15;
        int j = frag >> 3, k = frag & 7;               // 8 n-tiles x 8 k-tiles
        int col = j * 16 + (lane & 15);
        int row = k * 32 + (i >> 3) * 16 + (lane >> 4) * 8 + (i & 7);
        if (idx < 64 * 512) W1p[idx] = (_Float16)W1[row * 128 + col];
    }
    {
        int frag = idx >> 9, t = idx & 511;
        int lane = t >> 4, i = t & 15;
        int j = frag >> 2, k = frag & 3;               // 8 n-tiles x 4 k-tiles
        int col = j * 16 + (lane & 15);
        int row = k * 32 + (i >> 3) * 16 + (lane >> 4) * 8 + (i & 7);
        if (idx < 32 * 512) W2p[idx] = (_Float16)W2[row * 128 + col];
    }
}

__device__ __forceinline__ v16h cat8(v8h lo, v8h hi) {
    return __builtin_shufflevector(lo, hi, 0, 1, 2, 3, 4, 5, 6, 7,
                                   8, 9, 10, 11, 12, 13, 14, 15);
}

// ---------------- EdgeConv: fused 2-layer edge MLP with WMMA + atomic segment-max --------
// block = 128 threads (4 waves), 64 edges/block. Dyn LDS: sA 64x256 f16 (32K), sW1p 64K,
// sW2p 32K, sT 4x16x128 f16 (16K) => 147456 B.
__global__ __launch_bounds__(128) void edgeconv_kernel(
        const float* __restrict__ h2, const int* __restrict__ src, const int* __restrict__ dst,
        const _Float16* __restrict__ W1p, const float* __restrict__ b1,
        const _Float16* __restrict__ W2p, const float* __restrict__ b2,
        float* __restrict__ h3) {
    extern __shared__ _Float16 smem[];
    _Float16* sA  = smem;                      // [64][256]
    _Float16* sW1 = sA  + 64 * 256;            // packed: 64 frags * 512
    _Float16* sW2 = sW1 + 256 * 128;           // packed: 32 frags * 512
    _Float16* sT  = sW2 + 128 * 128;           // [4][16][128]
    __shared__ int sSrc[64], sDst[64];

    const int tid = threadIdx.x;
    const int e0  = blockIdx.x * 64;

    if (tid < 64) sSrc[tid] = src[e0 + tid];
    else          sDst[tid - 64] = dst[e0 + tid - 64];

    // stage packed weights with 16B copies
    {
        const uint4* g1 = (const uint4*)W1p;  uint4* s1 = (uint4*)sW1;
        for (int i = tid; i < 4096; i += 128) s1[i] = g1[i];
        const uint4* g2 = (const uint4*)W2p;  uint4* s2 = (uint4*)sW2;
        for (int i = tid; i < 2048; i += 128) s2[i] = g2[i];
    }
    __syncthreads();

    // stage edge features: [x_i, x_j - x_i] -> f16, float4-vectorized
    for (int idx = tid; idx < 64 * 64; idx += 128) {
        int el = idx >> 6, q = idx & 63;       // q = group of 4 of the 256 feature dims
        int di = sDst[el], si = sSrc[el];
        float4 v;
        if (q < 32) {
            v = *(const float4*)(h2 + di * 128 + q * 4);
        } else {
            float4 a = *(const float4*)(h2 + si * 128 + (q - 32) * 4);
            float4 b = *(const float4*)(h2 + di * 128 + (q - 32) * 4);
            v = make_float4(a.x - b.x, a.y - b.y, a.z - b.z, a.w - b.w);
        }
        v4h hv; hv[0] = (_Float16)v.x; hv[1] = (_Float16)v.y;
        hv[2] = (_Float16)v.z; hv[3] = (_Float16)v.w;
        *(v4h*)(sA + el * 256 + q * 4) = hv;
    }
    __syncthreads();

    const int wave = tid >> 5, lane = tid & 31;
    const int m = lane & 15, half = lane >> 4;
    const _Float16* rowA = sA + (wave * 16 + m) * 256;
    _Float16* tileT = sT + wave * 16 * 128;

    // ---- layer 1: [16x256] @ [256x128] + b1, relu -> sT (f16) ----
    v16h a1f[8];
    #pragma unroll
    for (int k = 0; k < 8; ++k) {
        const int ka = k * 32 + half * 8;
        a1f[k] = cat8(*(const v8h*)(rowA + ka), *(const v8h*)(rowA + ka + 16));
    }
    for (int j = 0; j < 8; ++j) {
        v8f acc = {0.f, 0.f, 0.f, 0.f, 0.f, 0.f, 0.f, 0.f};
        const int nb = j * 16 + m;
        const _Float16* wp = sW1 + ((j * 8) << 9) + (lane << 4);
        #pragma unroll
        for (int k = 0; k < 8; ++k) {
            v16h b = *(const v16h*)(wp + (k << 9));
            acc = __builtin_amdgcn_wmma_f32_16x16x32_f16(false, a1f[k], false, b,
                                                         (short)0, acc, false, false);
        }
        const float bias = b1[nb];
        #pragma unroll
        for (int i = 0; i < 8; ++i) {
            float v = acc[i] + bias;
            v = v > 0.f ? v : 0.f;
            tileT[(half * 8 + i) * 128 + nb] = (_Float16)v;
        }
    }
    __syncthreads();

    // ---- layer 2: [16x128] @ [128x128] + b2 -> atomic segment-max (relu folded) ----
    const _Float16* rowT = tileT + m * 128;
    v16h a2f[4];
    #pragma unroll
    for (int k = 0; k < 4; ++k) {
        const int ka = k * 32 + half * 8;
        a2f[k] = cat8(*(const v8h*)(rowT + ka), *(const v8h*)(rowT + ka + 16));
    }
    for (int j = 0; j < 8; ++j) {
        v8f acc = {0.f, 0.f, 0.f, 0.f, 0.f, 0.f, 0.f, 0.f};
        const int nb = j * 16 + m;
        const _Float16* wp = sW2 + ((j * 4) << 9) + (lane << 4);
        #pragma unroll
        for (int k = 0; k < 4; ++k) {
            v16h b = *(const v16h*)(wp + (k << 9));
            acc = __builtin_amdgcn_wmma_f32_16x16x32_f16(false, a2f[k], false, b,
                                                         (short)0, acc, false, false);
        }
        const float bias = b2[nb];
        #pragma unroll
        for (int i = 0; i < 8; ++i) {
            float v = acc[i] + bias;
            if (v > 0.f) {                         // result is relu(max(..)); init 0, skip <=0
                int d = sDst[wave * 16 + half * 8 + i];
                atomicMax((int*)(h3 + d * 128 + nb), __float_as_int(v));
            }
        }
    }
}

// ---------------- GCN4: aggregate 128-ch h3, then [N,128]@[128,256] ----------------
__global__ void gcn4_scatter(const float* __restrict__ h3, const int* __restrict__ src,
                             const int* __restrict__ dst, const float* __restrict__ dinv,
                             float* __restrict__ aggx) {
    int idx = blockIdx.x * blockDim.x + threadIdx.x;
    if (idx >= NE * 128) return;
    int e = idx >> 7, c = idx & 127;
    int s = src[e], d = dst[e];
    float nrm = dinv[s] * dinv[d];
    atomicAdd(&aggx[d * 128 + c], h3[s * 128 + c] * nrm);
}

__global__ void gcn4_fin(const float* __restrict__ aggx, const float* __restrict__ h3,
                         const float* __restrict__ dinv, const float* __restrict__ W,
                         const float* __restrict__ b, float* __restrict__ h4) {
    __shared__ float sz[128];
    int n = blockIdx.x, c = threadIdx.x;           // 256 threads
    if (c < 128) {
        float di = dinv[n];
        sz[c] = aggx[n * 128 + c] + h3[n * 128 + c] * di * di;
    }
    __syncthreads();
    float acc = b[c];
    #pragma unroll 8
    for (int k = 0; k < 128; ++k) acc += sz[k] * W[k * 256 + c];
    h4[n * 256 + c] = acc > 0.f ? acc : 0.f;
}

// ---------------- pooling ----------------
__global__ void pool_scatter(const float* __restrict__ h4, const int* __restrict__ batch,
                             float* __restrict__ gsum, float* __restrict__ gmax,
                             float* __restrict__ cnt) {
    int idx = blockIdx.x * blockDim.x + threadIdx.x;
    if (idx >= NN * 256) return;
    int n = idx >> 8, c = idx & 255;
    int g = batch[n];
    float v = h4[idx];                              // v >= 0 (post-relu)
    atomicAdd(&gsum[g * 256 + c], v);
    atomicMax((int*)(gmax + g * 256 + c), __float_as_int(v));
    if (c == 0) atomicAdd(&cnt[g], 1.0f);
}

// ---------------- head MLP + log_softmax ----------------
__global__ void mlp_kernel(const float* __restrict__ gsum, const float* __restrict__ gmax,
                           const float* __restrict__ cnt,
                           const float* __restrict__ W1, const float* __restrict__ b1,
                           const float* __restrict__ W2, const float* __restrict__ b2,
                           const float* __restrict__ W3, const float* __restrict__ b3,
                           float* __restrict__ out) {
    __shared__ float g[512], a1[128], a2[64];
    int gi = blockIdx.x, t = threadIdx.x;          // 128 threads
    float c = cnt[gi]; c = c > 1.f ? c : 1.f;
    for (int i = t; i < 256; i += 128) {
        g[i]       = gsum[gi * 256 + i] / c;
        g[256 + i] = gmax[gi * 256 + i];
    }
    __syncthreads();
    float acc = b1[t];
    for (int k = 0; k < 512; ++k) acc += g[k] * W1[k * 128 + t];
    a1[t] = acc > 0.f ? acc : 0.f;
    __syncthreads();
    if (t < 64) {
        float a = b2[t];
        for (int k = 0; k < 128; ++k) a += a1[k] * W2[k * 64 + t];
        a2[t] = a > 0.f ? a : 0.f;
    }
    __syncthreads();
    if (t == 0) {
        float o0 = b3[0], o1 = b3[1];
        for (int k = 0; k < 64; ++k) { o0 += a2[k] * W3[k * 2]; o1 += a2[k] * W3[k * 2 + 1]; }
        float mx = o0 > o1 ? o0 : o1;
        float lse = mx + logf(expf(o0 - mx) + expf(o1 - mx));
        out[gi * 2]     = o0 - lse;
        out[gi * 2 + 1] = o1 - lse;
    }
}

// ---------------- host launcher ----------------
extern "C" void kernel_launch(void* const* d_in, const int* in_sizes, int n_in,
                              void* d_out, int out_size, void* d_ws, size_t ws_size,
                              hipStream_t stream) {
    const float* x       = (const float*)d_in[0];
    const int*   ei      = (const int*)  d_in[1];
    const int*   batch   = (const int*)  d_in[2];
    const float* gcn1_W  = (const float*)d_in[3];
    const float* gcn1_b  = (const float*)d_in[4];
    const float* sage_Wl = (const float*)d_in[5];
    const float* sage_Wr = (const float*)d_in[6];
    const float* sage_b  = (const float*)d_in[7];
    const float* edge_W1 = (const float*)d_in[8];
    const float* edge_b1 = (const float*)d_in[9];
    const float* edge_W2 = (const float*)d_in[10];
    const float* edge_b2 = (const float*)d_in[11];
    const float* gcn4_W  = (const float*)d_in[12];
    const float* gcn4_b  = (const float*)d_in[13];
    const float* res_W   = (const float*)d_in[14];
    const float* res_b   = (const float*)d_in[15];
    const float* fc1_W   = (const float*)d_in[16];
    const float* fc1_b   = (const float*)d_in[17];
    const float* fc2_W   = (const float*)d_in[18];
    const float* fc2_b   = (const float*)d_in[19];
    const float* fc3_W   = (const float*)d_in[20];
    const float* fc3_b   = (const float*)d_in[21];
    float* out = (float*)d_out;

    const int* src = ei;
    const int* dst = ei + NE;

    // ---- workspace layout (256B aligned) ----
    char* ws = (char*)d_ws;
    size_t o = 0;
    auto take = [&](size_t bytes) { char* p = ws + o; o = (o + bytes + 255) & ~(size_t)255; return p; };
    float*    deg   = (float*)take(NN * 4);
    float*    dinv  = (float*)take(NN * 4);
    float*    aggx1 = (float*)take(NN * 2 * 4);
    float*    h1    = (float*)take((size_t)NN * 64 * 4);
    float*    ssum  = (float*)take((size_t)NN * 64 * 4);
    float*    h2    = (float*)take((size_t)NN * 128 * 4);
    float*    h3    = (float*)take((size_t)NN * 128 * 4);
    float*    aggx4 = (float*)take((size_t)NN * 128 * 4);
    float*    h4    = (float*)take((size_t)NN * 256 * 4);
    _Float16* W1p   = (_Float16*)take(256 * 128 * 2);
    _Float16* W2p   = (_Float16*)take(128 * 128 * 2);
    float*    gsum  = (float*)take(NG * 256 * 4);
    float*    gmaxb = (float*)take(NG * 256 * 4);
    float*    gcnt  = (float*)take(NG * 4);

    // ---- zero accumulators ----
    hipMemsetAsync(deg,   0, NN * 4, stream);
    hipMemsetAsync(aggx1, 0, NN * 2 * 4, stream);
    hipMemsetAsync(ssum,  0, (size_t)NN * 64 * 4, stream);
    hipMemsetAsync(h3,    0, (size_t)NN * 128 * 4, stream);
    hipMemsetAsync(aggx4, 0, (size_t)NN * 128 * 4, stream);
    hipMemsetAsync(gsum,  0, NG * 256 * 4, stream);
    hipMemsetAsync(gmaxb, 0, NG * 256 * 4, stream);
    hipMemsetAsync(gcnt,  0, NG * 4, stream);

    deg_kernel <<<(NE + 255) / 256, 256, 0, stream>>>(dst, deg);
    dinv_kernel<<<(NN + 255) / 256, 256, 0, stream>>>(deg, dinv);

    gcn1_scatter<<<(2 * NE + 255) / 256, 256, 0, stream>>>(x, src, dst, dinv, aggx1);
    gcn1_fin    <<<(NN * 64 + 255) / 256, 256, 0, stream>>>(x, aggx1, dinv, gcn1_W, gcn1_b, h1);

    sage_scatter<<<(size_t)NE * 64 / 256, 256, 0, stream>>>(h1, src, dst, ssum);
    sage_fin    <<<NN, 128, 0, stream>>>(h1, ssum, deg, sage_Wl, sage_Wr, res_W,
                                         sage_b, res_b, h2);

    cvt_w<<<(64 * 512 + 255) / 256, 256, 0, stream>>>(edge_W1, edge_W2, W1p, W2p);

    const size_t lds = (size_t)(64 * 256 + 256 * 128 + 128 * 128 + 4 * 16 * 128) * 2; // 147456 B
    hipFuncSetAttribute((const void*)edgeconv_kernel,
                        hipFuncAttributeMaxDynamicSharedMemorySize, (int)lds);
    edgeconv_kernel<<<NE / 64, 128, lds, stream>>>(h2, src, dst, W1p, edge_b1, W2p, edge_b2, h3);

    gcn4_scatter<<<(size_t)NE * 128 / 256, 256, 0, stream>>>(h3, src, dst, dinv, aggx4);
    gcn4_fin    <<<NN, 256, 0, stream>>>(aggx4, h3, dinv, gcn4_W, gcn4_b, h4);

    pool_scatter<<<(size_t)NN * 256 / 256, 256, 0, stream>>>(h4, batch, gsum, gmaxb, gcnt);
    mlp_kernel  <<<NG, 128, 0, stream>>>(gsum, gmaxb, gcnt, fc1_W, fc1_b, fc2_W, fc2_b,
                                         fc3_W, fc3_b, out);
}
